// GottliebKANLayer_34780645163401
// MI455X (gfx1250) — matrix operand order repaired and checked by reading, not compiled, not run on hardware
//
#include <hip/hip_runtime.h>
#include <stdint.h>

// Problem dims
#define B_ROWS 8192
#define I_DIM  1024
#define O_DIM  1024
#define DEG    5
#define ND     6                   // DEG+1
#define K_DIM  (I_DIM * ND)        // 6144
#define LN_EPS 1e-5f

// Tiling
#define BLK_M   32                 // rows per block (full N=1024 kept for fused LayerNorm)
#define KSTEP   32                 // WMMA K per step (bf16)
#define KSTEPS  (K_DIM / KSTEP)    // 192
#define ICHUNK  16                 // i per A-staging chunk -> 96 K = 3 ksteps
#define BPITCH  1040               // bf16 per padded B row in LDS (1024 + 16 pad)
#define BROWB   (BPITCH * 2)       // 2080 bytes per B row
#define BBUF    (32 * BROWB)       // 66560 bytes per B buffer
#define APITCH  104                // bf16 per padded A row (96 + 8 pad)
#define A_OFF   (2 * BBUF)         // A after double-buffered B
#define RED_OFF (A_OFF + BLK_M * APITCH * 2)
#define SMEM_BYTES (RED_OFF + 64 * 4)   // ~140 KB dynamic LDS

typedef __attribute__((ext_vector_type(16))) __bf16 bf16x16;
typedef __attribute__((ext_vector_type(8)))  float  f32x8;

__device__ __forceinline__ uint16_t f2bf(float f) {
    // round-to-nearest-even f32 -> bf16
    uint32_t u = __float_as_uint(f);
    u += 0x7FFFu + ((u >> 16) & 1u);
    return (uint16_t)(u >> 16);
}

__device__ __forceinline__ uint32_t pack2(float a, float b) {
    return (uint32_t)f2bf(a) | ((uint32_t)f2bf(b) << 16);
}

// ---------------------------------------------------------------------------
// Kernel 1: repack coeffs[I][O][6] f32 -> Bmat[K=I*6][N=O] bf16  (12.6 MB, L2-resident)
// ---------------------------------------------------------------------------
__global__ __launch_bounds__(256) void coeffs_to_bf16(
    const float* __restrict__ coeffs, uint16_t* __restrict__ Bmat)
{
    int idx = blockIdx.x * 256 + threadIdx.x;   // over K_DIM*O_DIM, exact multiple
    int k = idx >> 10;                          // / 1024
    int n = idx & 1023;
    int i = k / ND;
    int d = k - i * ND;
    Bmat[idx] = f2bf(coeffs[(i * O_DIM + n) * ND + d]);
}

// ---------------------------------------------------------------------------
// Kernel 2: fused basis + GEMM (bf16 WMMA) + LayerNorm
// grid = B_ROWS/BLK_M = 256 blocks, 256 threads (8 wave32 waves)
// wave w owns output columns [w*128, w*128+128)
// ---------------------------------------------------------------------------
__global__ __launch_bounds__(256) void kan_gemm_ln(
    const float* __restrict__ x, const float* __restrict__ alphap,
    const uint16_t* __restrict__ Bmat,
    const float* __restrict__ lnw, const float* __restrict__ lnb,
    float* __restrict__ out)
{
    extern __shared__ char smem[];
    const unsigned gss  = __builtin_amdgcn_groupstaticsize();
    const int tid   = threadIdx.x;
    const int lane  = tid & 31;
    const int wave  = tid >> 5;
    const int rowBase = blockIdx.x * BLK_M;
    const float alpha = alphap[0];

    f32x8 acc[2][8];
#pragma unroll
    for (int mt = 0; mt < 2; ++mt)
#pragma unroll
        for (int nt = 0; nt < 8; ++nt) acc[mt][nt] = (f32x8)0.0f;

    // ---- async staging of a 32x1024 bf16 B tile (64 KB) into LDS buffer `buf`
    auto issueB = [&](int ks, int buf) {
        const uint64_t gbase = (uint64_t)(uintptr_t)Bmat + (uint64_t)ks * 32u * 2048u;
        const unsigned lbase = gss + (unsigned)(buf * BBUF);
#pragma unroll
        for (int j = 0; j < 16; ++j) {
            int c  = tid + j * 256;        // 4096 x 16B chunks
            int r  = c >> 7;               // B row 0..31
            int co = (c & 127) << 4;       // byte col 0..2032
            unsigned laddr = lbase + (unsigned)(r * BROWB + co);
            uint64_t gaddr = gbase + (uint64_t)(r * 2048 + co);
            asm volatile("global_load_async_to_lds_b128 %0, %1, off"
                         :: "v"(laddr), "v"(gaddr) : "memory");
        }
    };

    // ---- compute sigmoid + Gottlieb basis for 32 rows x 16 inputs, store bf16 A tile
    auto stageA = [&](int ic) {
        const int i0 = ic * ICHUNK;
#pragma unroll
        for (int e = tid; e < BLK_M * ICHUNK; e += 256) {   // 2 iterations
            int r  = e >> 4;
            int ii = e & 15;
            float xv = x[(rowBase + r) * I_DIM + i0 + ii];
            float s  = 1.0f / (1.0f + __expf(-xv));
            float t0 = 1.0f;
            float t1 = 2.0f * alpha * s;
            float t2 = 2.0f * (alpha + 1.0f) * s * t1 - (alpha + 2.0f) * t0;
            float t3 = 2.0f * (alpha + 2.0f) * s * t2 - (alpha + 4.0f) * t1;
            float t4 = 2.0f * (alpha + 3.0f) * s * t3 - (alpha + 6.0f) * t2;
            float t5 = 2.0f * (alpha + 4.0f) * s * t4 - (alpha + 8.0f) * t3;
            // k-within-chunk = ii*6 + d ; 12 bytes, 4B aligned
            uint32_t* arow = (uint32_t*)(smem + A_OFF + r * (APITCH * 2) + ii * 12);
            arow[0] = pack2(t0, t1);
            arow[1] = pack2(t2, t3);
            arow[2] = pack2(t4, t5);
        }
    };

    issueB(0, 0);

    for (int ks = 0; ks < KSTEPS; ++ks) {
        const int buf = ks & 1;
        const int kk  = ks % 3;                  // kstep within i-chunk
        if (ks + 1 < KSTEPS) {
            issueB(ks + 1, buf ^ 1);             // prefetch next tile
            asm volatile("s_wait_asynccnt 16" ::: "memory");  // current tile done
        } else {
            asm volatile("s_wait_asynccnt 0" ::: "memory");
        }
        if (kk == 0) stageA(ks / 3);
        __syncthreads();   // all waves' async parts + A tile visible

        // ---- preload ALL fragments into distinct registers so the 4+128
        //      ds loads (A b128, B 16-bit) are in flight together and drain with partial
        //      s_wait_dscnt counts while WMMAs retire (no per-pair full waits).
        union Frag { uint4 q[2]; bf16x16 v; };

        // A fragments: lane L<16 -> row M=L, K {0..7,16..23}; L>=16 -> K {8..15,24..31}
        Frag afrag[2];
        const int arow0 = lane & 15;
        const int khalf = lane >> 4;
#pragma unroll
        for (int mt = 0; mt < 2; ++mt) {
            const char* p = smem + A_OFF
                          + ((mt * 16 + arow0) * APITCH + kk * 32 + khalf * 8) * 2;
            afrag[mt].q[0] = *(const uint4*)p;          // K = kk*32 + khalf*8 + 0..7
            afrag[mt].q[1] = *(const uint4*)(p + 32);   // K = +16
        }
        // B fragments: lane -> col N=L%16, K {0..7,16..23} (L<16) / {8..15,24..31} (L>=16)
        Frag bfrag[8];
        const uint16_t* bbase = (const uint16_t*)(smem + buf * BBUF) + wave * 128 + arow0;
#pragma unroll
        for (int nt = 0; nt < 8; ++nt) {
#pragma unroll
            for (int s = 0; s < 16; ++s) {
                int kb = khalf * 8 + (s & 7) + ((s >> 3) << 4);
                ((uint16_t*)&bfrag[nt])[s] = bbase[kb * BPITCH + nt * 16];
            }
        }
        // ---- 16 back-to-back WMMAs
#pragma unroll
        for (int nt = 0; nt < 8; ++nt) {
#pragma unroll
            for (int mt = 0; mt < 2; ++mt) {
                acc[mt][nt] = __builtin_amdgcn_wmma_f32_16x16x32_bf16(
                    false, afrag[mt].v, false, bfrag[nt].v,
                    (short)0, acc[mt][nt], false, false);
            }
        }
        __syncthreads();   // protect buf^1 / A from next round's overwrite
    }

    // ---- fused LayerNorm over the full 1024-wide stripe
    float* red = (float*)(smem + RED_OFF);     // [0..31]=sum, [32..63]=sumsq
    if (tid < 64) red[tid] = 0.0f;
    __syncthreads();

    const int rsel = (lane >> 4) << 3;         // D layout: lanes 16-31 hold M=r+8
#pragma unroll
    for (int mt = 0; mt < 2; ++mt) {
#pragma unroll
        for (int r = 0; r < 8; ++r) {
            float s1 = 0.0f, s2 = 0.0f;
#pragma unroll
            for (int nt = 0; nt < 8; ++nt) {
                float v = acc[mt][nt][r];
                s1 += v; s2 += v * v;
            }
            int row = mt * 16 + rsel + r;
            atomicAdd(&red[row], s1);          // ds_add_f32
            atomicAdd(&red[32 + row], s2);
        }
    }
    __syncthreads();

#pragma unroll
    for (int mt = 0; mt < 2; ++mt) {
#pragma unroll
        for (int nt = 0; nt < 8; ++nt) {
            int col = (wave * 8 + nt) * 16 + (lane & 15);
            float w = lnw[col], b = lnb[col];
#pragma unroll
            for (int r = 0; r < 8; ++r) {
                int row  = mt * 16 + rsel + r;
                float mu  = red[row] * (1.0f / O_DIM);
                float var = red[32 + row] * (1.0f / O_DIM) - mu * mu;
                float rstd = rsqrtf(var + LN_EPS);
                out[(size_t)(rowBase + row) * O_DIM + col] =
                    (acc[mt][nt][r] - mu) * rstd * w + b;
            }
        }
    }
}

// ---------------------------------------------------------------------------
extern "C" void kernel_launch(void* const* d_in, const int* in_sizes, int n_in,
                              void* d_out, int out_size, void* d_ws, size_t ws_size,
                              hipStream_t stream) {
    const float* x      = (const float*)d_in[0];
    const float* alpha  = (const float*)d_in[1];
    const float* coeffs = (const float*)d_in[2];
    const float* lnw    = (const float*)d_in[3];
    const float* lnb    = (const float*)d_in[4];
    float*       out    = (float*)d_out;
    uint16_t*    Bmat   = (uint16_t*)d_ws;     // 12.6 MB bf16, L2-resident

    hipLaunchKernelGGL(coeffs_to_bf16, dim3((K_DIM * O_DIM) / 256), dim3(256), 0, stream,
                       coeffs, Bmat);
    hipLaunchKernelGGL(kan_gemm_ln, dim3(B_ROWS / BLK_M), dim3(256), SMEM_BYTES, stream,
                       x, alpha, Bmat, lnw, lnb, out);
}
